// GraphConvolution_17824114278497
// MI455X (gfx1250) — compile-verified, matrix operand order checked by reading, not run on hardware
//
#include <hip/hip_runtime.h>
#include <math.h>

// ---------------------------------------------------------------------------
// GraphConvolution fused pipeline for MI455X (gfx1250, wave32, WMMA bf16)
//   scores = Xt Xt^T ; adj = softmax(scores) ; out = adj (Xt W)
// flash-attention formulation (Q=K=Xt, V=support). K/V tiles staged per-WG
// into LDS by the Tensor Data Mover (double buffered, TENSORcnt synced) so
// the kernel stays WMMA-bound instead of L2-bound.
// ---------------------------------------------------------------------------

typedef __attribute__((ext_vector_type(16))) __bf16 v16bf;
typedef __attribute__((ext_vector_type(8)))  __bf16 v8bf;
typedef __attribute__((ext_vector_type(8)))  float  v8f;
typedef __attribute__((ext_vector_type(4)))  unsigned int u32x4;
typedef __attribute__((ext_vector_type(8)))  int i32x8;
typedef __attribute__((ext_vector_type(4)))  int i32x4;

#define B_   4
#define C_   256
#define N_   4096
#define OUT_ 256
#define EPS_ 1e-5f
#define SLOPE_ 0.01f

union Frag { v16bf v; v8bf h[2]; };

// A-operand (16x32, row-major source) / B-operand (32x16, N-major source)
// fragment load: per-lane two aligned b128 chunks.
// lane L: row/col = L&15 ; K chunks at (L>>4)*8 and +16 (elements).
static __device__ __forceinline__ v16bf load_frag(const __bf16* p) {
  Frag f;
  f.h[0] = *(const v8bf*)(p);
  f.h[1] = *(const v8bf*)(p + 16);
  return f.v;
}

// Generic pointer to LDS byte offset: LDS aperture keeps the offset in the
// low 32 address bits (ISA 10.2 aperture mapping).
static __device__ __forceinline__ unsigned lds_off(const void* p) {
  return (unsigned)(uintptr_t)p;
}

// Issue a TDM 2D tile load (bf16 elements) via tensor_load_to_lds.
// D# packing per CDNA5 ISA ch.8: group0 = {flags, lds_addr, global_addr, type},
// group1 = {data_size, tensor_dim0/1, tile_dim0/1, dim0_stride}; groups 2/3
// zero (2D tensor). All operands are workgroup-uniform -> SGPRs.
static __device__ __forceinline__ void tdm_load_2d(
    unsigned lds_byte_off, const void* gaddr,
    unsigned tile_d0, unsigned tile_d1,       // tile extents (elements)
    unsigned tensor_d0, unsigned tensor_d1,   // tensor extents (elements)
    unsigned long long d0_stride)             // row stride (elements)
{
  unsigned long long ga = (unsigned long long)(uintptr_t)gaddr;
  u32x4 g0;
  g0[0] = 1u;                                  // count=1 (valid), user mode
  g0[1] = lds_byte_off;                        // lds_addr (bytes)
  g0[2] = (unsigned)(ga & 0xffffffffu);        // global_addr[31:0]
  g0[3] = (unsigned)((ga >> 32) & 0x1ffffffu)  // global_addr[56:32]
        | (2u << 30);                          // type = 2 ("image")
  i32x8 g1;
  g1[0] = (int)(1u << 16);                     // data_size=1 (2B), wg_mask=0
  g1[1] = (int)((tensor_d0 & 0xffffu) << 16);  // tensor_dim0[15:0] @ [63:48]
  g1[2] = (int)(((tensor_d0 >> 16) & 0xffffu)  // tensor_dim0[31:16]
        | ((tensor_d1 & 0xffffu) << 16));      // tensor_dim1[15:0]
  g1[3] = (int)(((tensor_d1 >> 16) & 0xffffu)  // tensor_dim1[31:16]
        | ((tile_d0 & 0xffffu) << 16));        // tile_dim0
  g1[4] = (int)(tile_d1 & 0xffffu);            // tile_dim1 ; tile_dim2=0
  g1[5] = (int)(d0_stride & 0xffffffffu);      // tensor_dim0_stride[31:0]
  g1[6] = (int)((d0_stride >> 32) & 0xffffu);  // stride[47:32]; dim1_stride=0
  g1[7] = 0;
  i32x4 gz = {0, 0, 0, 0};
#if __clang_major__ >= 23
  i32x8 gz8 = {0, 0, 0, 0, 0, 0, 0, 0};
  __builtin_amdgcn_tensor_load_to_lds(g0, g1, gz, gz, gz8, 0);
#else
  __builtin_amdgcn_tensor_load_to_lds(g0, g1, gz, gz, 0);
#endif
}

// ---------------- kernel 1a: x[B,C,N] -> xbf[B,N,C] (bf16) -----------------
__global__ void k_transpose_bf16(const float* __restrict__ x,
                                 __bf16* __restrict__ xbf) {
  __shared__ float tile[32][33];
  const int ntiles = N_ / 32;
  int bidx = blockIdx.x;
  int b  = bidx / ((C_ / 32) * ntiles);
  int r  = bidx % ((C_ / 32) * ntiles);
  int ct = r / ntiles;
  int nt = r % ntiles;
  int tx = threadIdx.x & 31;
  int ty = threadIdx.x >> 5;       // 0..7
  const float* xb = x + (size_t)b * C_ * N_;
  #pragma unroll
  for (int i = 0; i < 32; i += 8) {
    int c = ct * 32 + ty + i;
    int n = nt * 32 + tx;
    tile[ty + i][tx] = xb[(size_t)c * N_ + n];   // coalesced in n
  }
  __syncthreads();
  __bf16* xo = xbf + (size_t)b * N_ * C_;
  #pragma unroll
  for (int i = 0; i < 32; i += 8) {
    int n = nt * 32 + ty + i;
    int c = ct * 32 + tx;
    xo[(size_t)n * C_ + c] = (__bf16)tile[tx][ty + i];  // coalesced in c
  }
}

// ---------------- kernel 1b: Wt[o][c] = bf16(W[c][o]) ----------------------
__global__ void k_wt(const float* __restrict__ W, __bf16* __restrict__ Wt) {
  int idx = blockIdx.x * blockDim.x + threadIdx.x;
  if (idx >= C_ * OUT_) return;
  int c = idx / OUT_;
  int o = idx % OUT_;
  Wt[(size_t)o * C_ + c] = (__bf16)W[idx];
}

// -------- kernel 2: supT[b][o][n] = bf16( sum_c xbf[b][n][c] * W[c][o] ) ---
__global__ void __launch_bounds__(256)
k_support(const __bf16* __restrict__ xbf, const __bf16* __restrict__ Wt,
          __bf16* __restrict__ supT) {
  int lane = threadIdx.x & 31;
  int wave = threadIdx.x >> 5;
  int t = blockIdx.x * 8 + wave;            // one 16x16 tile per wave
  const int OT = OUT_ / 16;                 // 16
  const int NT = N_ / 16;                   // 256
  int b  = t / (NT * OT);
  int rr = t % (NT * OT);
  int nt = rr / OT;
  int ot = rr % OT;
  int l15 = lane & 15;
  int kb  = (lane >> 4) * 8;

  const __bf16* arow = xbf + (size_t)b * N_ * C_ + (size_t)(nt * 16 + l15) * C_ + kb;
  const __bf16* brow = Wt + (size_t)(ot * 16 + l15) * C_ + kb;
  v8f acc = {};
  #pragma unroll
  for (int kc = 0; kc < 8; ++kc) {          // K = 256 in chunks of 32
    v16bf a  = load_frag(arow + kc * 32);
    v16bf bm = load_frag(brow + kc * 32);
    acc = __builtin_amdgcn_wmma_f32_16x16x32_bf16(false, a, false, bm,
                                                  (short)0, acc, false, false);
  }
  // C/D layout: VGPR r -> row (n) = r + 8*(lane>=16), lane&15 -> col (o)
  __bf16* sb = supT + (size_t)b * OUT_ * N_;
  int o = ot * 16 + l15;
  int nbase = nt * 16 + ((lane >> 4) << 3);
  #pragma unroll
  for (int r = 0; r < 8; ++r)
    sb[(size_t)o * N_ + nbase + r] = (__bf16)acc[r];
}

// -------- kernel 3: flash attention + LeakyReLU -> y[b][o][n] (f32) --------
// WG = 256 threads = 8 waves: wave w -> rowTile (w>>1), OUT half (w&1).
// Per 32-column tile, wave 0 drives the TDM to stage K (32x256) and V
// (256x32) tiles into double-buffered LDS; all 8 waves consume them.
__global__ void __launch_bounds__(256)
k_flash(const __bf16* __restrict__ xbf, const __bf16* __restrict__ supT,
        float* __restrict__ y) {
  __shared__ __align__(16) __bf16 lds_k[2][32 * C_];    // 2 x 16KB
  __shared__ __align__(16) __bf16 lds_v[2][OUT_ * 32];  // 2 x 16KB
  __shared__ __align__(16) __bf16 lds_p[8][16 * 32];    // P bounce, 8KB
  int lane = threadIdx.x & 31;
  int wave = threadIdx.x >> 5;
  int b  = blockIdx.x >> 6;                // 64 WGs per batch
  int br = blockIdx.x & 63;
  int row_base = br * 64 + (wave >> 1) * 16;
  int o_base   = (wave & 1) * 128;
  int l15 = lane & 15;
  int g   = lane >> 4;
  int kb  = g * 8;
  const int MT = N_ / 32;

  const __bf16* xb = xbf + (size_t)b * N_ * C_;
  const __bf16* sb = supT + (size_t)b * OUT_ * N_;

  // Q fragments: 16 rows x C=256 (8 K-chunks), resident in VGPRs
  v16bf q[8];
  {
    const __bf16* qrow = xb + (size_t)(row_base + l15) * C_ + kb;
    #pragma unroll
    for (int kc = 0; kc < 8; ++kc) q[kc] = load_frag(qrow + kc * 32);
  }

  v8f zero = {};
  v8f oacc[8];
  #pragma unroll
  for (int i = 0; i < 8; ++i) oacc[i] = zero;
  float mrun[8], lrun[8];
  #pragma unroll
  for (int r = 0; r < 8; ++r) { mrun[r] = -__builtin_inff(); lrun[r] = 0.f; }

  __bf16* pbuf = &lds_p[wave][0];

  // prologue: TDM-stage tile 0 into buffer 0
  if (wave == 0) {
    tdm_load_2d(lds_off(&lds_k[0][0]), xb, C_, 32, C_, N_, C_);
    tdm_load_2d(lds_off(&lds_v[0][0]), sb, 32, OUT_, N_, OUT_, N_);
  }

  for (int mt = 0; mt < MT; ++mt) {
    int cur = mt & 1;
    if (wave == 0) {
      if (mt + 1 < MT) {  // prefetch next tile into the other buffer
        int m1 = (mt + 1) * 32;
        tdm_load_2d(lds_off(&lds_k[cur ^ 1][0]), xb + (size_t)m1 * C_,
                    C_, 32, C_, N_, C_);
        tdm_load_2d(lds_off(&lds_v[cur ^ 1][0]), sb + m1,
                    32, OUT_, N_, OUT_, N_);
        __builtin_amdgcn_s_wait_tensorcnt(2);  // current tile landed
      } else {
        __builtin_amdgcn_s_wait_tensorcnt(0);
      }
    }
    __syncthreads();  // publish staged tile to all waves

    const __bf16* kt = &lds_k[cur][0];
    const __bf16* vt = &lds_v[cur][0];

    // ---- scores tile S (16 x 32) = Q . K^T from LDS (ds_load_b128 frags)
    v8f s0 = {}, s1 = {};
    const __bf16* kr0 = kt + (size_t)l15 * C_ + kb;
    const __bf16* kr1 = kt + (size_t)(16 + l15) * C_ + kb;
    #pragma unroll
    for (int kc = 0; kc < 8; ++kc) {
      v16bf b0 = load_frag(kr0 + kc * 32);
      v16bf b1 = load_frag(kr1 + kc * 32);
      s0 = __builtin_amdgcn_wmma_f32_16x16x32_bf16(false, q[kc], false, b0,
                                                   (short)0, s0, false, false);
      s1 = __builtin_amdgcn_wmma_f32_16x16x32_bf16(false, q[kc], false, b1,
                                                   (short)0, s1, false, false);
    }

    // ---- online softmax (row stats per (VGPR r, lane-group g);
    //      columns are the 16 lanes of the group -> xor-shuffle reductions)
    float p0[8], p1[8];
    #pragma unroll
    for (int r = 0; r < 8; ++r) {
      float v = fmaxf(s0[r], s1[r]);
      v = fmaxf(v, __shfl_xor(v, 1, 32));
      v = fmaxf(v, __shfl_xor(v, 2, 32));
      v = fmaxf(v, __shfl_xor(v, 4, 32));
      v = fmaxf(v, __shfl_xor(v, 8, 32));
      float mnew  = fmaxf(mrun[r], v);
      float alpha = __expf(mrun[r] - mnew);
      mrun[r] = mnew;
      lrun[r] *= alpha;
      #pragma unroll
      for (int oc = 0; oc < 8; ++oc) oacc[oc][r] *= alpha;
      float e0 = __expf(s0[r] - mnew);
      float e1 = __expf(s1[r] - mnew);
      p0[r] = e0; p1[r] = e1;
      float s = e0 + e1;
      s += __shfl_xor(s, 1, 32);
      s += __shfl_xor(s, 2, 32);
      s += __shfl_xor(s, 4, 32);
      s += __shfl_xor(s, 8, 32);
      lrun[r] += s;
    }

    // ---- P: C/D layout -> A layout via wave-private LDS (same-wave DS order)
    #pragma unroll
    for (int r = 0; r < 8; ++r) {
      int row = r + (g << 3);
      pbuf[row * 32 + l15]      = (__bf16)p0[r];
      pbuf[row * 32 + 16 + l15] = (__bf16)p1[r];
    }
    v16bf pf = load_frag(pbuf + l15 * 32 + kb);

    // ---- O += P . V ; V tile in LDS, o-major rows of 32 (one K chunk)
    #pragma unroll
    for (int oc = 0; oc < 8; ++oc) {
      const __bf16* vrow = vt + (size_t)(o_base + oc * 16 + l15) * 32 + kb;
      v16bf vf = load_frag(vrow);
      oacc[oc] = __builtin_amdgcn_wmma_f32_16x16x32_bf16(false, pf, false, vf,
                                                         (short)0, oacc[oc], false, false);
    }

    __syncthreads();  // all waves done with buffers before next TDM overwrite
  }

  // ---- epilogue: 1/l, LeakyReLU, write y[b][o][n]
  float* yb = y + (size_t)b * OUT_ * N_;
  #pragma unroll
  for (int oc = 0; oc < 8; ++oc) {
    int o = o_base + oc * 16 + l15;
    #pragma unroll
    for (int r = 0; r < 8; ++r) {
      int n = row_base + (g << 3) + r;
      float v = oacc[oc][r] / lrun[r];
      v = (v >= 0.f) ? v : SLOPE_ * v;
      yb[(size_t)o * N_ + n] = v;
    }
  }
}

// -------- kernel 4: per-channel mean/var over (b, n) -----------------------
__global__ void k_bnstats(const float* __restrict__ y, float* __restrict__ stats) {
  __shared__ float ssum[256], ssq[256];
  int o = blockIdx.x;
  int tid = threadIdx.x;
  float s = 0.f, q = 0.f;
  for (int b = 0; b < B_; ++b) {
    const float* yo = y + (size_t)b * OUT_ * N_ + (size_t)o * N_;
    for (int n = tid; n < N_; n += 256) {
      float v = yo[n];
      s += v; q += v * v;
    }
  }
  ssum[tid] = s; ssq[tid] = q;
  __syncthreads();
  for (int off = 128; off > 0; off >>= 1) {
    if (tid < off) { ssum[tid] += ssum[tid + off]; ssq[tid] += ssq[tid + off]; }
    __syncthreads();
  }
  if (tid == 0) {
    float mean = ssum[0] / (float)(B_ * N_);
    float var  = ssq[0] / (float)(B_ * N_) - mean * mean;
    stats[o] = mean;
    stats[OUT_ + o] = var;
  }
}

// -------- kernel 5: normalize in place -------------------------------------
__global__ void k_bnapply(float* __restrict__ out, const float* __restrict__ stats,
                          const float* __restrict__ gamma, const float* __restrict__ beta) {
  size_t idx = (size_t)blockIdx.x * blockDim.x + threadIdx.x;
  const size_t total = (size_t)B_ * OUT_ * N_;
  if (idx >= total) return;
  int o = (int)((idx / N_) % OUT_);
  float v = out[idx];
  float r = rsqrtf(stats[OUT_ + o] + EPS_);
  out[idx] = gamma[o] * (v - stats[o]) * r + beta[o];
}

// ---------------------------------------------------------------------------
extern "C" void kernel_launch(void* const* d_in, const int* in_sizes, int n_in,
                              void* d_out, int out_size, void* d_ws, size_t ws_size,
                              hipStream_t stream) {
  const float* x     = (const float*)d_in[0];
  const float* W     = (const float*)d_in[1];
  const float* gamma = (const float*)d_in[2];
  const float* beta  = (const float*)d_in[3];
  float* y = (float*)d_out;

  char* ws = (char*)d_ws;
  const size_t xbf_bytes = (size_t)B_ * N_ * C_ * sizeof(__bf16);   // 8 MB
  const size_t sup_bytes = (size_t)B_ * OUT_ * N_ * sizeof(__bf16); // 8 MB
  __bf16* xbf  = (__bf16*)(ws);
  __bf16* supT = (__bf16*)(ws + xbf_bytes);
  __bf16* Wt   = (__bf16*)(ws + xbf_bytes + sup_bytes);
  float*  st   = (float*)(ws + xbf_bytes + sup_bytes + (size_t)C_ * OUT_ * sizeof(__bf16));

  (void)in_sizes; (void)n_in; (void)out_size; (void)ws_size;

  k_transpose_bf16<<<B_ * (C_ / 32) * (N_ / 32), 256, 0, stream>>>(x, xbf);
  k_wt<<<(C_ * OUT_ + 255) / 256, 256, 0, stream>>>(W, Wt);
  k_support<<<(B_ * (N_ / 16) * (OUT_ / 16)) / 8, 256, 0, stream>>>(xbf, Wt, supT);
  k_flash<<<B_ * (N_ / 64), 256, 0, stream>>>(xbf, supT, y);
  k_bnstats<<<OUT_, 256, 0, stream>>>(y, st);
  k_bnapply<<<(int)(((size_t)B_ * OUT_ * N_ + 255) / 256), 256, 0, stream>>>(y, st, gamma, beta);
}
